// GVPDrugEncoder_26362509263013
// MI455X (gfx1250) — matrix operand order repaired and checked by reading, not compiled, not run on hardware
//
#include <hip/hip_runtime.h>
#include <hip/hip_bf16.h>
#include <stdint.h>

#define N_NODES  50000
#define N_EDGES  200000
#define N_GRAPHS 2000
#define NPG      25
#define NSH      128
#define NVH      16

// ---------- WMMA plumbing (gfx1250, wave32, 16x16x32 bf16 -> f32) ----------
typedef __bf16        bf16x16 __attribute__((ext_vector_type(16)));
typedef float         f32x8   __attribute__((ext_vector_type(8)));
typedef unsigned int  u32x4   __attribute__((ext_vector_type(4)));

union BFrag { u32x4 u[2]; bf16x16 b; };
union AccU  { f32x8 v; float f[8]; };

__device__ __forceinline__ unsigned short f2bf(float f) {
  unsigned int x = __float_as_uint(f);
  return (unsigned short)((x + 0x7FFFu + ((x >> 16) & 1u)) >> 16);  // RNE
}
__device__ __forceinline__ float sigmoidf_(float x) { return 1.f / (1.f + expf(-x)); }
__device__ __forceinline__ f32x8 zero8() { f32x8 z = {0.f,0.f,0.f,0.f,0.f,0.f,0.f,0.f}; return z; }

// A tile: LDS, row-major [16][KP] bf16 (KP multiple of 32).
// W: global, transposed+padded bf16 [so][KP]; lane's B fragment = two 16-byte chunks, 32 bytes apart.
__device__ __forceinline__ f32x8 wmma_kloop(const unsigned short* Atile, int KP,
                                            const unsigned short* Wt, int n0,
                                            int ktiles, int lane, f32x8 acc) {
  const int m  = lane & 15;
  const int kh = lane >> 4;
  const unsigned short* arow = Atile + m * KP;
  const unsigned short* wl   = Wt + (size_t)(n0 + m) * KP;
  for (int kt = 0; kt < ktiles; ++kt) {
    const int kb = kt * 32;
    BFrag a, b;
    a.u[0] = *(const u32x4*)(arow + kb + kh * 8);        // K kb+kh*8 .. +7
    a.u[1] = *(const u32x4*)(arow + kb + 16 + kh * 8);   // K kb+16+kh*8 .. +7
    b.u[0] = *(const u32x4*)(wl + kb + kh * 8);          // K kb+kh*8 .. +7 @ col n
    b.u[1] = *(const u32x4*)(wl + kb + 16 + kh * 8);     // K kb+16+kh*8 .. +7
    acc = __builtin_amdgcn_wmma_f32_16x16x32_bf16(false, a.b, false, b.b,
                                                  (short)0, acc, false, false);
  }
  return acc;
}

// ---------- utility kernels ----------
__global__ void k_fill0(float* p, int n) {
  int i = blockIdx.x * blockDim.x + threadIdx.x;
  if (i < n) p[i] = 0.f;
}
__global__ void k_prep_w(const float* __restrict__ w, unsigned short* __restrict__ wt,
                         int din, int so, int dinpad) {
  int i = blockIdx.x * blockDim.x + threadIdx.x;
  if (i >= so * dinpad) return;
  int n = i / dinpad, k = i - n * dinpad;
  wt[i] = (k < din) ? f2bf(w[(size_t)k * so + n]) : (unsigned short)0;
}

// ---------- node_in GVP: (66 s, 1 vec) -> (128 s, 16 vec) ----------
__global__ __launch_bounds__(256) void k_node_in(
    const float* __restrict__ node_s, const float* __restrict__ node_v,
    const float* __restrict__ wh, const float* __restrict__ wsb,
    const unsigned short* __restrict__ wt, const float* __restrict__ wv,
    float* __restrict__ s_out, float* __restrict__ v_out) {
  __shared__ unsigned short A[16 * 96];
  const int tid = threadIdx.x;
  const int nb  = blockIdx.x * 16;
  for (int idx = tid; idx < 16 * 66; idx += 256) {
    int r = idx / 66, c = idx - r * 66;
    A[r * 96 + c] = f2bf(node_s[(size_t)(nb + r) * 66 + c]);
  }
  for (int idx = tid; idx < 16 * 14; idx += 256) {
    int r = idx / 14;
    A[r * 96 + 82 + (idx - r * 14)] = 0;
  }
  if (tid < 16) {
    int r = tid;
    float vx = node_v[(size_t)(nb + r) * 3 + 0];
    float vy = node_v[(size_t)(nb + r) * 3 + 1];
    float vz = node_v[(size_t)(nb + r) * 3 + 2];
    float vn2 = vx * vx + vy * vy + vz * vz;
    for (int h = 0; h < 16; ++h) {
      float w = wh[h];
      A[r * 96 + 66 + h] = f2bf(sqrtf(fmaxf(w * w * vn2, 1e-8f)));
    }
    for (int o = 0; o < 16; ++o) {
      float wvh = 0.f;
      for (int h = 0; h < 16; ++h) wvh += wh[h] * wv[h * 16 + o];
      float g = sigmoidf_(sqrtf(fmaxf(wvh * wvh * vn2, 1e-8f)));
      v_out[(size_t)(nb + r) * 48 + o * 3 + 0] = wvh * vx * g;
      v_out[(size_t)(nb + r) * 48 + o * 3 + 1] = wvh * vy * g;
      v_out[(size_t)(nb + r) * 48 + o * 3 + 2] = wvh * vz * g;
    }
  }
  __syncthreads();
  const int wave = tid >> 5, lane = tid & 31, kh = lane >> 4, nl = lane & 15;
  f32x8 acc = wmma_kloop(A, 96, wt, wave * 16, 3, lane, zero8());
  AccU o; o.v = acc;
  for (int r = 0; r < 8; ++r) {
    int m = r + 8 * kh, n = wave * 16 + nl;
    s_out[(size_t)(nb + m) * 128 + n] = fmaxf(o.f[r] + wsb[n], 0.f);
  }
}

// ---------- edge_in GVP: (16 s, 1 vec) -> (32 s, 1 vec), VALU ----------
__global__ __launch_bounds__(256) void k_edge_in(
    const float* __restrict__ edge_s, const float* __restrict__ edge_v,
    const float* __restrict__ wh, const float* __restrict__ wsb,
    const float* __restrict__ wsw, const float* __restrict__ wv,
    float* __restrict__ es, float* __restrict__ ev) {
  int e = blockIdx.x * blockDim.x + threadIdx.x;
  if (e >= N_EDGES) return;
  float sc[17];
  for (int i = 0; i < 16; ++i) sc[i] = edge_s[(size_t)e * 16 + i];
  float vx = edge_v[(size_t)e * 3 + 0], vy = edge_v[(size_t)e * 3 + 1], vz = edge_v[(size_t)e * 3 + 2];
  float vn2 = vx * vx + vy * vy + vz * vz;
  float w = wh[0];
  sc[16] = sqrtf(fmaxf(w * w * vn2, 1e-8f));
  for (int o = 0; o < 32; ++o) {
    float acc = wsb[o];
    for (int i = 0; i < 17; ++i) acc += sc[i] * wsw[i * 32 + o];
    es[(size_t)e * 32 + o] = fmaxf(acc, 0.f);
  }
  float c = w * wv[0];
  float g = sigmoidf_(sqrtf(fmaxf(c * c * vn2, 1e-8f)));
  ev[(size_t)e * 3 + 0] = c * vx * g;
  ev[(size_t)e * 3 + 1] = c * vy * g;
  ev[(size_t)e * 3 + 2] = c * vz * g;
}

// ---------- fused message kernel: msg0 -> msg1 -> msg2 -> scatter-add ----------
__global__ __launch_bounds__(256) void k_msg(
    const float* __restrict__ s, const float* __restrict__ v,
    const float* __restrict__ es, const float* __restrict__ ev,
    const int* __restrict__ eidx,
    const float* wh0, const float* wsb0, const unsigned short* wt0, const float* wv0,
    const float* wh1, const float* wsb1, const unsigned short* wt1, const float* wv1,
    const float* wh2, const float* wsb2, const unsigned short* wt2, const float* wv2,
    float* __restrict__ agg_s, float* __restrict__ agg_v, float* __restrict__ cnt) {
  __shared__ unsigned short A1[16 * 352];   // [s_src, es, s_dst, |vh|(33), pad] bf16
  __shared__ unsigned short A2[16 * 160];   // [ms, |vh'|(16), pad] bf16
  __shared__ float MV[16][99];              // concat vectors (33 x 3)
  __shared__ float VH[16][99];              // vh of msg0   (33 x 3)
  __shared__ float VB[16][48];              // vector state (16 x 3)
  __shared__ float VHB[16][48];             // vh of msg1/2 (16 x 3)
  __shared__ int SRC[16], DST[16];
  const int tid = threadIdx.x;
  const int eb  = blockIdx.x * 16;
  if (tid < 16) { SRC[tid] = eidx[eb + tid]; DST[tid] = eidx[N_EDGES + eb + tid]; }
  __syncthreads();
  for (int idx = tid; idx < 16 * 288; idx += 256) {
    int r = idx / 288, c = idx - r * 288;
    float val;
    if (c < 128)      val = s[(size_t)SRC[r] * 128 + c];
    else if (c < 160) val = es[(size_t)(eb + r) * 32 + (c - 128)];
    else              val = s[(size_t)DST[r] * 128 + (c - 160)];
    A1[r * 352 + c] = f2bf(val);
  }
  for (int idx = tid; idx < 16 * 31; idx += 256) {
    int r = idx / 31;
    A1[r * 352 + 321 + (idx - r * 31)] = 0;
  }
  for (int idx = tid; idx < 16 * 99; idx += 256) {
    int r = idx / 99, j = idx - r * 99, i = j / 3, c = j - i * 3;
    float val;
    if (i < 16)       val = v[(size_t)SRC[r] * 48 + i * 3 + c];
    else if (i == 16) val = ev[(size_t)(eb + r) * 3 + c];
    else              val = v[(size_t)DST[r] * 48 + (i - 17) * 3 + c];
    MV[r][j] = val;
  }
  __syncthreads();
  // vh = MV(33,3) x wh0(33,33); norms -> A1 cols 288..320
  for (int idx = tid; idx < 16 * 33; idx += 256) {
    int r = idx / 33, h = idx - r * 33;
    float a0 = 0, a1 = 0, a2 = 0;
    for (int i = 0; i < 33; ++i) {
      float w = wh0[i * 33 + h];
      a0 += MV[r][i * 3 + 0] * w; a1 += MV[r][i * 3 + 1] * w; a2 += MV[r][i * 3 + 2] * w;
    }
    VH[r][h * 3 + 0] = a0; VH[r][h * 3 + 1] = a1; VH[r][h * 3 + 2] = a2;
    A1[r * 352 + 288 + h] = f2bf(sqrtf(fmaxf(a0 * a0 + a1 * a1 + a2 * a2, 1e-8f)));
  }
  __syncthreads();
  // v1 = VH(33,3) x wv0(33,16), gated
  for (int idx = tid; idx < 256; idx += 256) {
    int r = idx >> 4, o = idx & 15;
    float a0 = 0, a1 = 0, a2 = 0;
    for (int h = 0; h < 33; ++h) {
      float w = wv0[h * 16 + o];
      a0 += VH[r][h * 3 + 0] * w; a1 += VH[r][h * 3 + 1] * w; a2 += VH[r][h * 3 + 2] * w;
    }
    float g = sigmoidf_(sqrtf(fmaxf(a0 * a0 + a1 * a1 + a2 * a2, 1e-8f)));
    VB[r][o * 3 + 0] = a0 * g; VB[r][o * 3 + 1] = a1 * g; VB[r][o * 3 + 2] = a2 * g;
  }
  __syncthreads();
  const int wave = tid >> 5, lane = tid & 31, kh = lane >> 4, nl = lane & 15;
  // msg0 GEMM (K=352)
  f32x8 acc = wmma_kloop(A1, 352, wt0, wave * 16, 11, lane, zero8());
  { AccU o; o.v = acc;
    for (int r = 0; r < 8; ++r) {
      int m = r + 8 * kh, n = wave * 16 + nl;
      A2[m * 160 + n] = f2bf(fmaxf(o.f[r] + wsb0[n], 0.f));
    } }
  // vh1 = VB(16,3) x wh1(16,16); norms -> A2 cols 128..143; pad 144..159
  for (int idx = tid; idx < 256; idx += 256) {
    int r = idx >> 4, h = idx & 15;
    float a0 = 0, a1 = 0, a2 = 0;
    for (int i = 0; i < 16; ++i) {
      float w = wh1[i * 16 + h];
      a0 += VB[r][i * 3 + 0] * w; a1 += VB[r][i * 3 + 1] * w; a2 += VB[r][i * 3 + 2] * w;
    }
    VHB[r][h * 3 + 0] = a0; VHB[r][h * 3 + 1] = a1; VHB[r][h * 3 + 2] = a2;
    A2[r * 160 + 128 + h] = f2bf(sqrtf(fmaxf(a0 * a0 + a1 * a1 + a2 * a2, 1e-8f)));
    A2[r * 160 + 144 + h] = 0;
  }
  __syncthreads();
  // msg1 GEMM (K=160)
  acc = wmma_kloop(A2, 160, wt1, wave * 16, 5, lane, zero8());
  __syncthreads();   // all A2 reads done before overwrite
  { AccU o; o.v = acc;
    for (int r = 0; r < 8; ++r) {
      int m = r + 8 * kh, n = wave * 16 + nl;
      A2[m * 160 + n] = f2bf(fmaxf(o.f[r] + wsb1[n], 0.f));
    } }
  // v2 = VHB x wv1, gated -> VB
  for (int idx = tid; idx < 256; idx += 256) {
    int r = idx >> 4, o = idx & 15;
    float a0 = 0, a1 = 0, a2 = 0;
    for (int h = 0; h < 16; ++h) {
      float w = wv1[h * 16 + o];
      a0 += VHB[r][h * 3 + 0] * w; a1 += VHB[r][h * 3 + 1] * w; a2 += VHB[r][h * 3 + 2] * w;
    }
    float g = sigmoidf_(sqrtf(fmaxf(a0 * a0 + a1 * a1 + a2 * a2, 1e-8f)));
    VB[r][o * 3 + 0] = a0 * g; VB[r][o * 3 + 1] = a1 * g; VB[r][o * 3 + 2] = a2 * g;
  }
  __syncthreads();
  // vh2 = VB x wh2; norms -> A2 cols 128..143
  for (int idx = tid; idx < 256; idx += 256) {
    int r = idx >> 4, h = idx & 15;
    float a0 = 0, a1 = 0, a2 = 0;
    for (int i = 0; i < 16; ++i) {
      float w = wh2[i * 16 + h];
      a0 += VB[r][i * 3 + 0] * w; a1 += VB[r][i * 3 + 1] * w; a2 += VB[r][i * 3 + 2] * w;
    }
    VHB[r][h * 3 + 0] = a0; VHB[r][h * 3 + 1] = a1; VHB[r][h * 3 + 2] = a2;
    A2[r * 160 + 128 + h] = f2bf(sqrtf(fmaxf(a0 * a0 + a1 * a1 + a2 * a2, 1e-8f)));
  }
  __syncthreads();
  // msg2 GEMM (K=160, no relu) -> scatter-add to destination nodes
  acc = wmma_kloop(A2, 160, wt2, wave * 16, 5, lane, zero8());
  { AccU o; o.v = acc;
    for (int r = 0; r < 8; ++r) {
      int m = r + 8 * kh, n = wave * 16 + nl;
      atomicAdd(&agg_s[(size_t)DST[m] * 128 + n], o.f[r] + wsb2[n]);
    } }
  // v3 = VHB x wv2 (no gate) -> scatter-add
  for (int idx = tid; idx < 256; idx += 256) {
    int r = idx >> 4, o = idx & 15;
    float a0 = 0, a1 = 0, a2 = 0;
    for (int h = 0; h < 16; ++h) {
      float w = wv2[h * 16 + o];
      a0 += VHB[r][h * 3 + 0] * w; a1 += VHB[r][h * 3 + 1] * w; a2 += VHB[r][h * 3 + 2] * w;
    }
    int d = DST[r];
    atomicAdd(&agg_v[(size_t)d * 48 + o * 3 + 0], a0);
    atomicAdd(&agg_v[(size_t)d * 48 + o * 3 + 1], a1);
    atomicAdd(&agg_v[(size_t)d * 48 + o * 3 + 2], a2);
  }
  if (tid < 16) atomicAdd(&cnt[DST[tid]], 1.f);
}

// ---------- fused node update: residual + LN + ff0 + ff1 + residual + LN ----------
__global__ __launch_bounds__(256) void k_node_update(
    float* __restrict__ s, float* __restrict__ v,
    const float* __restrict__ agg_s, const float* __restrict__ agg_v,
    const float* __restrict__ cnt,
    const float* whf0, const float* bf0, const unsigned short* wtf0, const float* wvf0,
    const float* whf1, const float* bf1, const unsigned short* wtf1, const float* wvf1) {
  __shared__ unsigned short A0[16 * 160];
  __shared__ unsigned short AB[16 * 544];
  __shared__ float SR[16][128];
  __shared__ float VR[16][48];
  __shared__ float VH[16][96];
  __shared__ float VF[16][96];
  __shared__ float VH2[16][96];
  __shared__ float STAT[16][4];   // mu, rstd, vnorm
  const int tid = threadIdx.x;
  const int nb  = blockIdx.x * 16;
  for (int idx = tid; idx < 16 * 128; idx += 256) {
    int r = idx >> 7, c = idx & 127;
    float cn = fmaxf(cnt[nb + r], 1.f);
    SR[r][c] = s[(size_t)(nb + r) * 128 + c] + agg_s[(size_t)(nb + r) * 128 + c] / cn;
  }
  for (int idx = tid; idx < 16 * 48; idx += 256) {
    int r = idx / 48, c = idx - r * 48;
    float cn = fmaxf(cnt[nb + r], 1.f);
    VR[r][c] = v[(size_t)(nb + r) * 48 + c] + agg_v[(size_t)(nb + r) * 48 + c] / cn;
  }
  __syncthreads();
  if (tid < 16) {
    int r = tid; float mu = 0.f;
    for (int c = 0; c < 128; ++c) mu += SR[r][c];
    mu *= (1.f / 128.f);
    float var = 0.f;
    for (int c = 0; c < 128; ++c) { float d = SR[r][c] - mu; var += d * d; }
    STAT[r][0] = mu; STAT[r][1] = rsqrtf(var * (1.f / 128.f) + 1e-5f);
  } else if (tid < 32) {
    int r = tid - 16; float acc = 0.f;
    for (int h = 0; h < 16; ++h) {
      float a = VR[r][h * 3], b = VR[r][h * 3 + 1], c = VR[r][h * 3 + 2];
      acc += fmaxf(a * a + b * b + c * c, 1e-8f);
    }
    STAT[r][2] = sqrtf(acc * (1.f / 16.f));
  }
  __syncthreads();
  for (int idx = tid; idx < 16 * 128; idx += 256) {
    int r = idx >> 7, c = idx & 127;
    float val = (SR[r][c] - STAT[r][0]) * STAT[r][1];
    SR[r][c] = val;
    A0[r * 160 + c] = f2bf(val);
  }
  for (int idx = tid; idx < 16 * 48; idx += 256) {
    int r = idx / 48, c = idx - r * 48;
    VR[r][c] = VR[r][c] / STAT[r][2];
  }
  __syncthreads();
  // vh(ff0) = VR(16,3) x whf0(16,32); norms -> A0 cols 128..159
  for (int idx = tid; idx < 16 * 32; idx += 256) {
    int r = idx >> 5, h = idx & 31;
    float a0 = 0, a1 = 0, a2 = 0;
    for (int i = 0; i < 16; ++i) {
      float w = whf0[i * 32 + h];
      a0 += VR[r][i * 3] * w; a1 += VR[r][i * 3 + 1] * w; a2 += VR[r][i * 3 + 2] * w;
    }
    VH[r][h * 3] = a0; VH[r][h * 3 + 1] = a1; VH[r][h * 3 + 2] = a2;
    A0[r * 160 + 128 + h] = f2bf(sqrtf(fmaxf(a0 * a0 + a1 * a1 + a2 * a2, 1e-8f)));
  }
  __syncthreads();
  const int wave = tid >> 5, lane = tid & 31, kh = lane >> 4, nl = lane & 15;
  // ff0 GEMM: 16x160 x 160x512 (each wave does 4 N-tiles)
  for (int t = 0; t < 4; ++t) {
    f32x8 acc = wmma_kloop(A0, 160, wtf0, wave * 16 + t * 128, 5, lane, zero8());
    AccU o; o.v = acc;
    for (int r = 0; r < 8; ++r) {
      int m = r + 8 * kh, n = wave * 16 + t * 128 + nl;
      AB[m * 544 + n] = f2bf(fmaxf(o.f[r] + bf0[n], 0.f));
    }
  }
  // vf = VH(32,3) x wvf0(32,32), gated
  for (int idx = tid; idx < 16 * 32; idx += 256) {
    int r = idx >> 5, o = idx & 31;
    float a0 = 0, a1 = 0, a2 = 0;
    for (int h = 0; h < 32; ++h) {
      float w = wvf0[h * 32 + o];
      a0 += VH[r][h * 3] * w; a1 += VH[r][h * 3 + 1] * w; a2 += VH[r][h * 3 + 2] * w;
    }
    float g = sigmoidf_(sqrtf(fmaxf(a0 * a0 + a1 * a1 + a2 * a2, 1e-8f)));
    VF[r][o * 3] = a0 * g; VF[r][o * 3 + 1] = a1 * g; VF[r][o * 3 + 2] = a2 * g;
  }
  __syncthreads();
  // vh2 = VF x whf1(32,32); norms -> AB cols 512..543
  for (int idx = tid; idx < 16 * 32; idx += 256) {
    int r = idx >> 5, h = idx & 31;
    float a0 = 0, a1 = 0, a2 = 0;
    for (int i = 0; i < 32; ++i) {
      float w = whf1[i * 32 + h];
      a0 += VF[r][i * 3] * w; a1 += VF[r][i * 3 + 1] * w; a2 += VF[r][i * 3 + 2] * w;
    }
    VH2[r][h * 3] = a0; VH2[r][h * 3 + 1] = a1; VH2[r][h * 3 + 2] = a2;
    AB[r * 544 + 512 + h] = f2bf(sqrtf(fmaxf(a0 * a0 + a1 * a1 + a2 * a2, 1e-8f)));
  }
  __syncthreads();
  // ff1 GEMM: 16x544 x 544x128 (no relu), residual into SR
  f32x8 acc = wmma_kloop(AB, 544, wtf1, wave * 16, 17, lane, zero8());
  { AccU o; o.v = acc;
    for (int r = 0; r < 8; ++r) {
      int m = r + 8 * kh, n = wave * 16 + nl;
      SR[m][n] += o.f[r] + bf1[n];
    } }
  // v residual: VR += VH2 x wvf1(32,16)
  for (int idx = tid; idx < 256; idx += 256) {
    int r = idx >> 4, o = idx & 15;
    float a0 = 0, a1 = 0, a2 = 0;
    for (int h = 0; h < 32; ++h) {
      float w = wvf1[h * 16 + o];
      a0 += VH2[r][h * 3] * w; a1 += VH2[r][h * 3 + 1] * w; a2 += VH2[r][h * 3 + 2] * w;
    }
    VR[r][o * 3] += a0; VR[r][o * 3 + 1] += a1; VR[r][o * 3 + 2] += a2;
  }
  __syncthreads();
  if (tid < 16) {
    int r = tid; float mu = 0.f;
    for (int c = 0; c < 128; ++c) mu += SR[r][c];
    mu *= (1.f / 128.f);
    float var = 0.f;
    for (int c = 0; c < 128; ++c) { float d = SR[r][c] - mu; var += d * d; }
    STAT[r][0] = mu; STAT[r][1] = rsqrtf(var * (1.f / 128.f) + 1e-5f);
  } else if (tid < 32) {
    int r = tid - 16; float acc2 = 0.f;
    for (int h = 0; h < 16; ++h) {
      float a = VR[r][h * 3], b = VR[r][h * 3 + 1], c = VR[r][h * 3 + 2];
      acc2 += fmaxf(a * a + b * b + c * c, 1e-8f);
    }
    STAT[r][2] = sqrtf(acc2 * (1.f / 16.f));
  }
  __syncthreads();
  for (int idx = tid; idx < 16 * 128; idx += 256) {
    int r = idx >> 7, c = idx & 127;
    s[(size_t)(nb + r) * 128 + c] = (SR[r][c] - STAT[r][0]) * STAT[r][1];
  }
  for (int idx = tid; idx < 16 * 48; idx += 256) {
    int r = idx / 48, c = idx - r * 48;
    v[(size_t)(nb + r) * 48 + c] = VR[r][c] / STAT[r][2];
  }
}

// ---------- final LN + out_proj (144->128, no act) -> dense ----------
__global__ __launch_bounds__(256) void k_out_proj(
    const float* __restrict__ s, const float* __restrict__ v,
    const float* wh, const float* wsb, const unsigned short* wt,
    float* __restrict__ dense) {
  __shared__ unsigned short A[16 * 160];
  __shared__ float SR[16][128];
  __shared__ float VR[16][48];
  __shared__ float STAT[16][4];
  const int tid = threadIdx.x;
  const int nb  = blockIdx.x * 16;
  for (int idx = tid; idx < 16 * 128; idx += 256) {
    int r = idx >> 7, c = idx & 127;
    SR[r][c] = s[(size_t)(nb + r) * 128 + c];
  }
  for (int idx = tid; idx < 16 * 48; idx += 256) {
    int r = idx / 48, c = idx - r * 48;
    VR[r][c] = v[(size_t)(nb + r) * 48 + c];
  }
  __syncthreads();
  if (tid < 16) {
    int r = tid; float mu = 0.f;
    for (int c = 0; c < 128; ++c) mu += SR[r][c];
    mu *= (1.f / 128.f);
    float var = 0.f;
    for (int c = 0; c < 128; ++c) { float d = SR[r][c] - mu; var += d * d; }
    STAT[r][0] = mu; STAT[r][1] = rsqrtf(var * (1.f / 128.f) + 1e-5f);
  } else if (tid < 32) {
    int r = tid - 16; float acc = 0.f;
    for (int h = 0; h < 16; ++h) {
      float a = VR[r][h * 3], b = VR[r][h * 3 + 1], c = VR[r][h * 3 + 2];
      acc += fmaxf(a * a + b * b + c * c, 1e-8f);
    }
    STAT[r][2] = sqrtf(acc * (1.f / 16.f));
  }
  __syncthreads();
  for (int idx = tid; idx < 16 * 128; idx += 256) {
    int r = idx >> 7, c = idx & 127;
    A[r * 160 + c] = f2bf((SR[r][c] - STAT[r][0]) * STAT[r][1]);
  }
  for (int idx = tid; idx < 16 * 48; idx += 256) {
    int r = idx / 48, c = idx - r * 48;
    VR[r][c] = VR[r][c] / STAT[r][2];
  }
  __syncthreads();
  for (int idx = tid; idx < 256; idx += 256) {
    int r = idx >> 4, h = idx & 15;
    float a0 = 0, a1 = 0, a2 = 0;
    for (int i = 0; i < 16; ++i) {
      float w = wh[i * 16 + h];
      a0 += VR[r][i * 3] * w; a1 += VR[r][i * 3 + 1] * w; a2 += VR[r][i * 3 + 2] * w;
    }
    A[r * 160 + 128 + h] = f2bf(sqrtf(fmaxf(a0 * a0 + a1 * a1 + a2 * a2, 1e-8f)));
    A[r * 160 + 144 + h] = 0;
  }
  __syncthreads();
  const int wave = tid >> 5, lane = tid & 31, kh = lane >> 4, nl = lane & 15;
  f32x8 acc = wmma_kloop(A, 160, wt, wave * 16, 5, lane, zero8());
  AccU o; o.v = acc;
  for (int r = 0; r < 8; ++r) {
    int m = r + 8 * kh, n = wave * 16 + nl;
    dense[(size_t)(nb + m) * 128 + n] = o.f[r] + wsb[n];
  }
}

__global__ void k_mask(float* mask) {
  int i = blockIdx.x * blockDim.x + threadIdx.x;
  if (i < N_NODES) mask[i] = 1.f;
}
__global__ void k_pool(const float* __restrict__ dense, float* __restrict__ pooled) {
  int i = blockIdx.x * blockDim.x + threadIdx.x;
  if (i >= N_GRAPHS * 128) return;
  int g = i >> 7, c = i & 127;
  float acc = 0.f;
  for (int p = 0; p < NPG; ++p) acc += dense[(size_t)(g * NPG + p) * 128 + c];
  pooled[(size_t)g * 128 + c] = acc * (1.f / (float)NPG);
}

// ---------------------------- host orchestration ----------------------------
extern "C" void kernel_launch(void* const* d_in, const int* in_sizes, int n_in,
                              void* d_out, int out_size, void* d_ws, size_t ws_size,
                              hipStream_t stream) {
  const float* node_s = (const float*)d_in[0];
  const float* node_v = (const float*)d_in[1];
  const float* edge_s = (const float*)d_in[2];
  const float* edge_v = (const float*)d_in[3];
  const int*   eidx   = (const int*)d_in[4];
  // d_in[5] = batch (identity layout: node i -> graph i/25, pos i%25; unused)
  // params flattened as jax pytree leaves: sorted dict keys at each level:
  //   edge_in(wh,ws_b,ws_w,wv), layers[0..2]{ff[0],ff[1],msg[0..2] each (wh,ws_b,ws_w,wv)},
  //   node_in(wh,ws_b,ws_w,wv), out_proj(wh,ws_b,ws_w)
  const float* ei_wh  = (const float*)d_in[6];
  const float* ei_wsb = (const float*)d_in[7];
  const float* ei_wsw = (const float*)d_in[8];
  const float* ei_wv  = (const float*)d_in[9];
  struct GVPW { const float *wh, *wsb, *wsw, *wv; };
  GVPW ff0[3], ff1[3], msg0[3], msg1[3], msg2[3];
  for (int l = 0; l < 3; ++l) {
    int B = 10 + 20 * l;
    ff0[l]  = { (const float*)d_in[B+0],  (const float*)d_in[B+1],  (const float*)d_in[B+2],  (const float*)d_in[B+3]  };
    ff1[l]  = { (const float*)d_in[B+4],  (const float*)d_in[B+5],  (const float*)d_in[B+6],  (const float*)d_in[B+7]  };
    msg0[l] = { (const float*)d_in[B+8],  (const float*)d_in[B+9],  (const float*)d_in[B+10], (const float*)d_in[B+11] };
    msg1[l] = { (const float*)d_in[B+12], (const float*)d_in[B+13], (const float*)d_in[B+14], (const float*)d_in[B+15] };
    msg2[l] = { (const float*)d_in[B+16], (const float*)d_in[B+17], (const float*)d_in[B+18], (const float*)d_in[B+19] };
  }
  const float* ni_wh  = (const float*)d_in[70];
  const float* ni_wsb = (const float*)d_in[71];
  const float* ni_wsw = (const float*)d_in[72];
  const float* ni_wv  = (const float*)d_in[73];
  const float* op_wh  = (const float*)d_in[74];
  const float* op_wsb = (const float*)d_in[75];
  const float* op_wsw = (const float*)d_in[76];

  // workspace carve
  char* wsb_ = (char*)d_ws; size_t off = 0;
  auto carve = [&](size_t bytes) -> void* {
    off = (off + 255) & ~(size_t)255;
    void* p = wsb_ + off; off += bytes; return p;
  };
  float* s     = (float*)carve((size_t)N_NODES * 128 * 4);
  float* v     = (float*)carve((size_t)N_NODES * 48 * 4);
  float* es    = (float*)carve((size_t)N_EDGES * 32 * 4);
  float* ev    = (float*)carve((size_t)N_EDGES * 3 * 4);
  float* agg_s = (float*)carve((size_t)N_NODES * 128 * 4);
  float* agg_v = (float*)carve((size_t)N_NODES * 48 * 4);
  float* cnt   = (float*)carve((size_t)N_NODES * 4);
  unsigned short* wt_ni  = (unsigned short*)carve((size_t)128 * 96 * 2);
  unsigned short* wt_op  = (unsigned short*)carve((size_t)128 * 160 * 2);
  unsigned short* wt_m0[3], *wt_m1[3], *wt_m2[3], *wt_f0[3], *wt_f1[3];
  for (int l = 0; l < 3; ++l) {
    wt_m0[l] = (unsigned short*)carve((size_t)128 * 352 * 2);
    wt_m1[l] = (unsigned short*)carve((size_t)128 * 160 * 2);
    wt_m2[l] = (unsigned short*)carve((size_t)128 * 160 * 2);
    wt_f0[l] = (unsigned short*)carve((size_t)512 * 160 * 2);
    wt_f1[l] = (unsigned short*)carve((size_t)128 * 544 * 2);
  }

  auto prep = [&](const float* w, unsigned short* wt, int din, int so, int dinpad) {
    int total = so * dinpad;
    k_prep_w<<<(total + 255) / 256, 256, 0, stream>>>(w, wt, din, so, dinpad);
  };
  prep(ni_wsw, wt_ni, 82, 128, 96);
  prep(op_wsw, wt_op, 144, 128, 160);
  for (int l = 0; l < 3; ++l) {
    prep(msg0[l].wsw, wt_m0[l], 321, 128, 352);
    prep(msg1[l].wsw, wt_m1[l], 144, 128, 160);
    prep(msg2[l].wsw, wt_m2[l], 144, 128, 160);
    prep(ff0[l].wsw,  wt_f0[l], 160, 512, 160);
    prep(ff1[l].wsw,  wt_f1[l], 544, 128, 544);
  }

  k_node_in<<<N_NODES / 16, 256, 0, stream>>>(node_s, node_v, ni_wh, ni_wsb, wt_ni, ni_wv, s, v);
  k_edge_in<<<(N_EDGES + 255) / 256, 256, 0, stream>>>(edge_s, edge_v, ei_wh, ei_wsb, ei_wsw, ei_wv, es, ev);

  for (int l = 0; l < 3; ++l) {
    k_fill0<<<(N_NODES * 128 + 255) / 256, 256, 0, stream>>>(agg_s, N_NODES * 128);
    k_fill0<<<(N_NODES * 48 + 255) / 256, 256, 0, stream>>>(agg_v, N_NODES * 48);
    k_fill0<<<(N_NODES + 255) / 256, 256, 0, stream>>>(cnt, N_NODES);
    k_msg<<<N_EDGES / 16, 256, 0, stream>>>(
        s, v, es, ev, eidx,
        msg0[l].wh, msg0[l].wsb, wt_m0[l], msg0[l].wv,
        msg1[l].wh, msg1[l].wsb, wt_m1[l], msg1[l].wv,
        msg2[l].wh, msg2[l].wsb, wt_m2[l], msg2[l].wv,
        agg_s, agg_v, cnt);
    k_node_update<<<N_NODES / 16, 256, 0, stream>>>(
        s, v, agg_s, agg_v, cnt,
        ff0[l].wh, ff0[l].wsb, wt_f0[l], ff0[l].wv,
        ff1[l].wh, ff1[l].wsb, wt_f1[l], ff1[l].wv);
  }

  float* out   = (float*)d_out;
  float* dense = out;                                        // 2000*25*128
  float* mask  = out + (size_t)N_GRAPHS * NPG * 128;         // 50000
  float* pool  = mask + N_NODES;                             // 2000*128
  k_out_proj<<<N_NODES / 16, 256, 0, stream>>>(s, v, op_wh, op_wsb, wt_op, dense);
  k_mask<<<(N_NODES + 255) / 256, 256, 0, stream>>>(mask);
  k_pool<<<(N_GRAPHS * 128 + 255) / 256, 256, 0, stream>>>(dense, pool);
  (void)in_sizes; (void)n_in; (void)out_size; (void)ws_size;
}
